// EGNN_58136677319440
// MI455X (gfx1250) — matrix operand order repaired and checked by reading, not compiled, not run on hardware
//
#include <hip/hip_runtime.h>

// ---------------------------------------------------------------------------
// EGNN layer forward for MI455X (gfx1250), wave32 + WMMA f16 (f32 accum).
// B=4, N=2048, K=32, d=64, E=31. All GEMMs mapped onto v_wmma_f32_16x16x32_f16.
// Weights pre-swizzled into per-lane B-fragment layout (1 vector load / frag).
// Branch-free erf-GELU epilogue (hw v_exp), 4-accumulator WMMA inner loops.
// ---------------------------------------------------------------------------

typedef __attribute__((ext_vector_type(16))) _Float16 v16h;
typedef __attribute__((ext_vector_type(8)))  float    v8f;

#define WMMA_F16(a, b, c) \
  __builtin_amdgcn_wmma_f32_16x16x32_f16(false, (a), false, (b), (short)0, (c), false, false)

// Branch-free exact-form GELU: erf via Abramowitz-Stegun 7.1.26 (|err|<1.5e-7),
// using the hardware exp path. No exec-mask divergence in the GEMM epilogues.
__device__ __forceinline__ float gelu_fast(float x) {
  float ax = fabsf(x) * 0.70710678118654752f;
  float t  = 1.0f / (1.0f + 0.3275911f * ax);
  float poly = t * (0.254829592f +
               t * (-0.284496736f +
               t * (1.421413741f +
               t * (-1.453152027f +
               t * 1.061405429f))));
  float erf_ax = 1.0f - poly * __expf(-ax * ax);
  float erfv = copysignf(erf_ax, x);
  return 0.5f * x * (1.0f + erfv);
}

__device__ __forceinline__ float sigmoid_fast(float x) {
  return 1.0f / (1.0f + __expf(-x));
}

// A fragment (16x32, f16, row-major source with leading dim ld):
// lanes 0-15: M=lane, elems 0-7 = K 0..7, elems 8-15 = K 16..23
// lanes16-31: M=lane-16, elems 0-7 = K 8..15, elems 8-15 = K 24..31
__device__ __forceinline__ v16h load_a_frag(const _Float16* base, int ld, int lane) {
  int m  = lane & 15;
  int kb = (lane < 16) ? 0 : 8;
  const _Float16* row = base + m * ld + kb;
  v16h a;
#pragma unroll
  for (int e = 0; e < 8; ++e) a[e] = row[e];
#pragma unroll
  for (int e = 0; e < 8; ++e) a[8 + e] = row[16 + e];
  return a;
}

// B fragment from swizzled weights: tile = 32 lanes x 16 contiguous halfs.
// Tile order: (nt * KC + kc) * 512 halfs. 32-byte aligned per lane.
__device__ __forceinline__ v16h load_b_frag_swz(const _Float16* tile, int lane) {
  return *((const v16h*)tile + lane);
}

// ---------------------------------------------------------------------------
// Kernel 0: convert fp32 weights -> f16, swizzled into WMMA B-fragment layout.
// For W[Ktot x Ntot] (Ktot mult of 32, Ntot mult of 16), tile (kc,nt) holds
//   lane n|kb, elem e  ->  W[kc*32 + kb + (e&7) + 2*(e&8)][nt*16 + n], kb=0/8   (k >= Krows -> 0)
// ws layout (halfs): We1h[49152] | We2h[16384] | Wc1h[16384] |
//                    Wn1h[32768] | Wn2h[16384] | node_in[8192*128]
// ---------------------------------------------------------------------------
__device__ __forceinline__ void swizzle_one(int idx, const float* __restrict__ W,
                                            int Krows, int Ktot, int Ntot,
                                            _Float16* __restrict__ dst) {
  int tile   = idx >> 9;        // 512 halfs per 32x16 tile
  int within = idx & 511;
  int lane   = within >> 4;
  int e      = within & 15;
  int KC = Ktot >> 5;
  int kc = tile % KC;
  int nt = tile / KC;
  int n  = nt * 16 + (lane & 15);
  int kb = (lane < 16) ? 0 : 8;
  int k  = kc * 32 + kb + (e & 7) + ((e & 8) << 1);
  float v = (k < Krows) ? W[k * Ntot + n] : 0.0f;
  dst[idx] = (_Float16)v;
}

__global__ __launch_bounds__(256) void convert_weights_kernel(
    const float* __restrict__ We1, const float* __restrict__ We2,
    const float* __restrict__ Wc1, const float* __restrict__ Wn1,
    const float* __restrict__ Wn2, _Float16* __restrict__ ws) {
  int idx = blockIdx.x * 256 + threadIdx.x;  // 512*256 = 131072 threads exactly
  if (idx < 49152) { swizzle_one(idx, We1, 168, 192, 256, ws); return; }
  int i2 = idx - 49152;
  if (i2 < 16384) { swizzle_one(i2, We2, 256, 256, 64, ws + 49152); return; }
  int i3 = i2 - 16384;
  if (i3 < 16384) { swizzle_one(i3, Wc1, 64, 64, 256, ws + 65536); return; }
  int i4 = i3 - 16384;
  if (i4 < 32768) { swizzle_one(i4, Wn1, 128, 128, 256, ws + 81920); return; }
  int i5 = i4 - 32768;
  if (i5 < 16384) { swizzle_one(i5, Wn2, 256, 256, 64, ws + 114688); return; }
}

// ---------------------------------------------------------------------------
// Kernel 1: one block per node (b,i). 256 threads = 8 waves.
// Fused: gather -> edge MLP -> gate -> coors MLP -> vw -> coord update,
//        m_i sum, LayerNorm(feats) -> stage node_in (f16) for kernel 2.
// ---------------------------------------------------------------------------
__global__ __launch_bounds__(256) void egnn_edge_kernel(
    const float* __restrict__ feats, const float* __restrict__ space,
    const float* __restrict__ edges, const int* __restrict__ E_idx,
    const _Float16* __restrict__ We1h, const float* __restrict__ be1,
    const _Float16* __restrict__ We2h, const float* __restrict__ be2,
    const float* __restrict__ Wg, const float* __restrict__ bg,
    const _Float16* __restrict__ Wc1h, const float* __restrict__ bc1,
    const float* __restrict__ Wc2, const float* __restrict__ bc2,
    const float* __restrict__ ln_g, const float* __restrict__ ln_b,
    const float* __restrict__ coors_scale,
    _Float16* __restrict__ node_in, float* __restrict__ space_out) {
  __shared__ _Float16 s_ein[32][192];  // edge MLP input tile (padded K=192)
  __shared__ _Float16 s_h[32][256];    // hidden activations (reused per stage)
  __shared__ float    s_m[32][64];     // gated m_ij (fp32)
  __shared__ _Float16 s_mh[32][64];    // gated m_ij (f16, A of coors GEMM)
  __shared__ float    s_fi[64];
  __shared__ _Float16 s_fih[64];
  __shared__ float    s_vn[32][3];     // CoorsNorm'd relative vectors
  __shared__ float    s_vw[32];
  __shared__ float    s_spi[3];
  __shared__ int      s_jidx[32];
  __shared__ float    s_mu, s_rstd;

  const int tid  = threadIdx.x;
  const int lane = tid & 31;
  const int wave = tid >> 5;
  const int bi   = blockIdx.x;   // = b*2048 + i
  const int b    = bi >> 11;

  if (tid < 64) {
    float v = feats[(size_t)bi * 64 + tid];
    s_fi[tid]  = v;
    s_fih[tid] = (_Float16)v;
  }
  if (tid < 3) s_spi[tid] = space[(size_t)bi * 3 + tid];
  if (tid >= 64 && tid < 96) s_jidx[tid - 64] = E_idx[(size_t)bi * 32 + (tid - 64)];
  __syncthreads();

  // LayerNorm statistics over feats_i (wave 0)
  if (wave == 0) {
    float v0 = s_fi[lane], v1 = s_fi[lane + 32];
    float s = v0 + v1;
#pragma unroll
    for (int o = 16; o > 0; o >>= 1) s += __shfl_xor(s, o, 32);
    float mu = s * (1.0f / 64.0f);
    float d0 = v0 - mu, d1 = v1 - mu;
    float vv = d0 * d0 + d1 * d1;
#pragma unroll
    for (int o = 16; o > 0; o >>= 1) vv += __shfl_xor(vv, o, 32);
    if (lane == 0) { s_mu = mu; s_rstd = rsqrtf(vv * (1.0f / 64.0f) + 1e-5f); }
  }

  // Build edge_input tile: [feats_i | feats_j | fourier(9) | edges(31) | pad(24)]
  for (int idx = tid; idx < 32 * 64; idx += 256) {
    int k = idx >> 6, c = idx & 63;
    s_ein[k][c] = s_fih[c];
    int j = s_jidx[k];
    s_ein[k][64 + c] = (_Float16)feats[((size_t)b * 2048 + j) * 64 + c];
  }
  if (tid < 32) {
    int k = tid, j = s_jidx[k];
    const float* sj = space + ((size_t)b * 2048 + j) * 3;
    float vx = sj[0] - s_spi[0], vy = sj[1] - s_spi[1], vz = sj[2] - s_spi[2];
    float e2  = vx * vx + vy * vy + vz * vz;
    float nrm = sqrtf(e2);
    float inv = coors_scale[0] / fmaxf(nrm, 1e-8f);
    s_vn[k][0] = vx * inv; s_vn[k][1] = vy * inv; s_vn[k][2] = vz * inv;
    float sc = 1.0f;
#pragma unroll
    for (int e = 0; e < 4; ++e) {
      float xs = e2 / sc; sc *= 2.0f;
      s_ein[k][128 + e] = (_Float16)__sinf(xs);
      s_ein[k][132 + e] = (_Float16)__cosf(xs);
    }
    s_ein[k][136] = (_Float16)e2;
  }
  const float* eg = edges + (size_t)bi * (32 * 31);
  for (int idx = tid; idx < 32 * 31; idx += 256) {
    int k = idx / 31, e = idx - k * 31;
    s_ein[k][137 + e] = (_Float16)eg[idx];
  }
  for (int idx = tid; idx < 32 * 24; idx += 256) {
    int k = idx / 24, c = idx - k * 24;
    s_ein[k][168 + c] = (_Float16)0.0f;
  }
  __syncthreads();

  // GEMM1: [32x192] @ We1[192x256] + be1 -> GELU -> s_h (f16).  KC = 6.
  // kc outer; 4 live accumulators (2 M-tiles x 2 N-tiles); each A/B frag once.
  {
    v8f a00 = {}, a01 = {}, a10 = {}, a11 = {};   // acc[m][t]
#pragma unroll
    for (int kc = 0; kc < 6; ++kc) {
      v16h fa0 = load_a_frag(&s_ein[0][0] + kc * 32, 192, lane);
      v16h fa1 = load_a_frag(&s_ein[16][0] + kc * 32, 192, lane);
      v16h fb0 = load_b_frag_swz(We1h + ((size_t)wave * 6 + kc) * 512, lane);
      v16h fb1 = load_b_frag_swz(We1h + ((size_t)(wave + 8) * 6 + kc) * 512, lane);
      a00 = WMMA_F16(fa0, fb0, a00);
      a10 = WMMA_F16(fa1, fb0, a10);
      a01 = WMMA_F16(fa0, fb1, a01);
      a11 = WMMA_F16(fa1, fb1, a11);
    }
    int nA = wave * 16 + (lane & 15);
    int nB = (wave + 8) * 16 + (lane & 15);
    int mb = (lane < 16) ? 0 : 8;
    float bvA = be1[nA], bvB = be1[nB];
#pragma unroll
    for (int r = 0; r < 8; ++r) {
      s_h[mb + r][nA]      = (_Float16)gelu_fast(a00[r] + bvA);
      s_h[16 + mb + r][nA] = (_Float16)gelu_fast(a10[r] + bvA);
      s_h[mb + r][nB]      = (_Float16)gelu_fast(a01[r] + bvB);
      s_h[16 + mb + r][nB] = (_Float16)gelu_fast(a11[r] + bvB);
    }
  }
  __syncthreads();

  // GEMM2: [32x256] @ We2[256x64] + be2 -> raw m_ij (fp32).  KC = 8.
  {
    int m0 = (wave & 1) * 16, nt = wave >> 1;
    int n0 = nt * 16;
    v8f acc = {};
#pragma unroll
    for (int kc = 0; kc < 8; ++kc) {
      v16h a  = load_a_frag(&s_h[m0][0] + kc * 32, 256, lane);
      v16h bf = load_b_frag_swz(We2h + ((size_t)nt * 8 + kc) * 512, lane);
      acc = WMMA_F16(a, bf, acc);
    }
    int n = n0 + (lane & 15);
    int mb = m0 + ((lane < 16) ? 0 : 8);
    float bv = be2[n];
#pragma unroll
    for (int r = 0; r < 8; ++r) s_m[mb + r][n] = acc[r] + bv;
  }
  __syncthreads();

  // Gate: m_ij *= sigmoid(m_ij @ Wg + bg)  (4 rows per wave, shuffle-reduced)
  {
    float wg0 = Wg[lane], wg1 = Wg[lane + 32];
    float bgv = bg[0];
#pragma unroll
    for (int rr = 0; rr < 4; ++rr) {
      int r = wave * 4 + rr;
      float m0v = s_m[r][lane], m1v = s_m[r][lane + 32];
      float p = m0v * wg0 + m1v * wg1;
#pragma unroll
      for (int o = 16; o > 0; o >>= 1) p += __shfl_xor(p, o, 32);
      float g = sigmoid_fast(p + bgv);
      float v0 = m0v * g, v1 = m1v * g;
      s_m[r][lane] = v0;              s_m[r][lane + 32] = v1;
      s_mh[r][lane] = (_Float16)v0;   s_mh[r][lane + 32] = (_Float16)v1;
    }
  }
  __syncthreads();

  // GEMM3 (coors MLP-1): [32x64] @ Wc1[64x256] + bc1 -> GELU -> s_h.  KC = 2.
  {
    v8f a00 = {}, a01 = {}, a10 = {}, a11 = {};
#pragma unroll
    for (int kc = 0; kc < 2; ++kc) {
      v16h fa0 = load_a_frag(&s_mh[0][0] + kc * 32, 64, lane);
      v16h fa1 = load_a_frag(&s_mh[16][0] + kc * 32, 64, lane);
      v16h fb0 = load_b_frag_swz(Wc1h + ((size_t)wave * 2 + kc) * 512, lane);
      v16h fb1 = load_b_frag_swz(Wc1h + ((size_t)(wave + 8) * 2 + kc) * 512, lane);
      a00 = WMMA_F16(fa0, fb0, a00);
      a10 = WMMA_F16(fa1, fb0, a10);
      a01 = WMMA_F16(fa0, fb1, a01);
      a11 = WMMA_F16(fa1, fb1, a11);
    }
    int nA = wave * 16 + (lane & 15);
    int nB = (wave + 8) * 16 + (lane & 15);
    int mb = (lane < 16) ? 0 : 8;
    float bvA = bc1[nA], bvB = bc1[nB];
#pragma unroll
    for (int r = 0; r < 8; ++r) {
      s_h[mb + r][nA]      = (_Float16)gelu_fast(a00[r] + bvA);
      s_h[16 + mb + r][nA] = (_Float16)gelu_fast(a10[r] + bvA);
      s_h[mb + r][nB]      = (_Float16)gelu_fast(a01[r] + bvB);
      s_h[16 + mb + r][nB] = (_Float16)gelu_fast(a11[r] + bvB);
    }
  }
  __syncthreads();

  // coors MLP-2: vw[k] = h2[k] . Wc2 + bc2  (4 rows per wave)
  {
    float bcv = bc2[0];
#pragma unroll
    for (int rr = 0; rr < 4; ++rr) {
      int r = wave * 4 + rr;
      float p = 0.0f;
#pragma unroll
      for (int t = 0; t < 8; ++t) {
        int c = lane + 32 * t;
        p += (float)s_h[r][c] * Wc2[c];
      }
#pragma unroll
      for (int o = 16; o > 0; o >>= 1) p += __shfl_xor(p, o, 32);
      if (lane == 0) s_vw[r] = p + bcv;
    }
  }
  __syncthreads();

  // Coordinate update: space_out = sum_k vw[k]*vn[k] + space_i
  if (wave == 0) {
    float w  = s_vw[lane];
    float cx = w * s_vn[lane][0], cy = w * s_vn[lane][1], cz = w * s_vn[lane][2];
#pragma unroll
    for (int o = 16; o > 0; o >>= 1) {
      cx += __shfl_xor(cx, o, 32);
      cy += __shfl_xor(cy, o, 32);
      cz += __shfl_xor(cz, o, 32);
    }
    if (lane == 0) {
      space_out[(size_t)bi * 3 + 0] = s_spi[0] + cx;
      space_out[(size_t)bi * 3 + 1] = s_spi[1] + cy;
      space_out[(size_t)bi * 3 + 2] = s_spi[2] + cz;
    }
  }
  // m_i = sum_k m_ij ; stage node_in = [LayerNorm(feats_i) | m_i] as f16
  if (tid < 64) {
    int c = tid;
    float mi = 0.0f;
#pragma unroll
    for (int r = 0; r < 32; ++r) mi += s_m[r][c];
    float normed = (s_fi[c] - s_mu) * s_rstd * ln_g[c] + ln_b[c];
    _Float16* ni = node_in + (size_t)bi * 128;
    ni[c]      = (_Float16)normed;
    ni[64 + c] = (_Float16)mi;
  }
}

// ---------------------------------------------------------------------------
// Kernel 2: node MLP, 32 rows per block. [32x128]@Wn1 -> GELU -> @Wn2 + feats
// ---------------------------------------------------------------------------
__global__ __launch_bounds__(256) void egnn_node_kernel(
    const _Float16* __restrict__ node_in,
    const _Float16* __restrict__ Wn1h, const float* __restrict__ bn1,
    const _Float16* __restrict__ Wn2h, const float* __restrict__ bn2,
    const float* __restrict__ feats, float* __restrict__ out_nodes) {
  __shared__ _Float16 s_a[32][128];
  __shared__ _Float16 s_h[32][256];
  const int tid = threadIdx.x, lane = tid & 31, wave = tid >> 5;
  const size_t row0 = (size_t)blockIdx.x * 32;

  for (int idx = tid; idx < 32 * 128; idx += 256) {
    int r = idx >> 7, c = idx & 127;
    s_a[r][c] = node_in[(row0 + r) * 128 + c];
  }
  __syncthreads();

  // GEMM1: [32x128] @ Wn1[128x256] + bn1 -> GELU -> s_h.  KC = 4.
  {
    v8f a00 = {}, a01 = {}, a10 = {}, a11 = {};
#pragma unroll
    for (int kc = 0; kc < 4; ++kc) {
      v16h fa0 = load_a_frag(&s_a[0][0] + kc * 32, 128, lane);
      v16h fa1 = load_a_frag(&s_a[16][0] + kc * 32, 128, lane);
      v16h fb0 = load_b_frag_swz(Wn1h + ((size_t)wave * 4 + kc) * 512, lane);
      v16h fb1 = load_b_frag_swz(Wn1h + ((size_t)(wave + 8) * 4 + kc) * 512, lane);
      a00 = WMMA_F16(fa0, fb0, a00);
      a10 = WMMA_F16(fa1, fb0, a10);
      a01 = WMMA_F16(fa0, fb1, a01);
      a11 = WMMA_F16(fa1, fb1, a11);
    }
    int nA = wave * 16 + (lane & 15);
    int nB = (wave + 8) * 16 + (lane & 15);
    int mb = (lane < 16) ? 0 : 8;
    float bvA = bn1[nA], bvB = bn1[nB];
#pragma unroll
    for (int r = 0; r < 8; ++r) {
      s_h[mb + r][nA]      = (_Float16)gelu_fast(a00[r] + bvA);
      s_h[16 + mb + r][nA] = (_Float16)gelu_fast(a10[r] + bvA);
      s_h[mb + r][nB]      = (_Float16)gelu_fast(a01[r] + bvB);
      s_h[16 + mb + r][nB] = (_Float16)gelu_fast(a11[r] + bvB);
    }
  }
  __syncthreads();

  // GEMM2: [32x256] @ Wn2[256x64] + bn2 + residual.  KC = 8.
  {
    int m0 = (wave & 1) * 16, nt = wave >> 1;
    int n0 = nt * 16;
    v8f acc = {};
#pragma unroll
    for (int kc = 0; kc < 8; ++kc) {
      v16h a  = load_a_frag(&s_h[m0][0] + kc * 32, 256, lane);
      v16h bf = load_b_frag_swz(Wn2h + ((size_t)nt * 8 + kc) * 512, lane);
      acc = WMMA_F16(a, bf, acc);
    }
    int n = n0 + (lane & 15);
    int mb = m0 + ((lane < 16) ? 0 : 8);
    float bv = bn2[n];
#pragma unroll
    for (int r = 0; r < 8; ++r) {
      size_t g = row0 + mb + r;
      out_nodes[g * 64 + n] = acc[r] + bv + feats[g * 64 + n];
    }
  }
}

// ---------------------------------------------------------------------------
extern "C" void kernel_launch(void* const* d_in, const int* in_sizes, int n_in,
                              void* d_out, int out_size, void* d_ws, size_t ws_size,
                              hipStream_t stream) {
  const float* feats = (const float*)d_in[0];
  const float* space = (const float*)d_in[1];
  const float* edges = (const float*)d_in[2];
  const int*   E_idx = (const int*)d_in[3];
  const float* We1 = (const float*)d_in[4];
  const float* be1 = (const float*)d_in[5];
  const float* We2 = (const float*)d_in[6];
  const float* be2 = (const float*)d_in[7];
  const float* Wg  = (const float*)d_in[8];
  const float* bg  = (const float*)d_in[9];
  const float* Wc1 = (const float*)d_in[10];
  const float* bc1 = (const float*)d_in[11];
  const float* Wc2 = (const float*)d_in[12];
  const float* bc2 = (const float*)d_in[13];
  const float* Wn1 = (const float*)d_in[14];
  const float* bn1 = (const float*)d_in[15];
  const float* Wn2 = (const float*)d_in[16];
  const float* bn2 = (const float*)d_in[17];
  const float* ln_g = (const float*)d_in[18];
  const float* ln_b = (const float*)d_in[19];
  const float* coors_scale = (const float*)d_in[20];

  _Float16* wsh = (_Float16*)d_ws;
  _Float16* We1h    = wsh + 0;       // 6*16 tiles  (K=192 pad, N=256)
  _Float16* We2h    = wsh + 49152;   // 8*4 tiles   (K=256, N=64)
  _Float16* Wc1h    = wsh + 65536;   // 2*16 tiles  (K=64,  N=256)
  _Float16* Wn1h    = wsh + 81920;   // 4*16 tiles  (K=128, N=256)
  _Float16* Wn2h    = wsh + 114688;  // 8*4 tiles   (K=256, N=64)
  _Float16* node_in = wsh + 131072;  // 8192*128

  float* out_nodes = (float*)d_out;
  float* out_space = out_nodes + (size_t)4 * 2048 * 64;

  hipLaunchKernelGGL(convert_weights_kernel, dim3(512), dim3(256), 0, stream,
                     We1, We2, Wc1, Wn1, Wn2, wsh);
  hipLaunchKernelGGL(egnn_edge_kernel, dim3(4 * 2048), dim3(256), 0, stream,
                     feats, space, edges, E_idx, We1h, be1, We2h, be2, Wg, bg,
                     Wc1h, bc1, Wc2, bc2, ln_g, ln_b, coors_scale,
                     node_in, out_space);
  hipLaunchKernelGGL(egnn_node_kernel, dim3(256), dim3(256), 0, stream,
                     node_in, Wn1h, bn1, Wn2h, bn2, feats, out_nodes);
}
